// StateSpace_6262062318268
// MI455X (gfx1250) — compile-verified, matrix operand order checked by reading, not run on hardware
//
#include <hip/hip_runtime.h>

// ---------------------------------------------------------------------------
// Fused encoder-RNN + state-space rollout for MI455X (gfx1250, wave32, WMMA).
//   K_IN=64, K_OUT=64, K_STATE=256, T_PRE=64, T_FWD=80, BATCH=4096
// Each block owns a 16-row batch tile and runs all 144 time steps locally.
// 16 waves/block, one 16-col state N-tile per wave. All weight fragments are
// preloaded into REGISTERS (bf16) once; only the h/input A-fragments stream
// from LDS each step, so v_wmma issue is the critical path. tanh uses the
// CDNA5 hardware V_TANH_F32 transcendental (co-executes with WMMA).
// ---------------------------------------------------------------------------

#define K_IN    64
#define K_OUT   64
#define K_STATE 256
#define T_PRE   64
#define T_FWD   80
#define BATCH   4096
#define BM      16       // batch rows per block (one WMMA M-tile)
#define NTHREADS 512     // 16 waves
#define HSTR    264      // h row stride in bf16 elems (256 + pad, 16B aligned)

typedef __bf16 v16bf __attribute__((ext_vector_type(16)));
typedef float  v8f   __attribute__((ext_vector_type(8)));

union FragAB { v16bf v; uint4 q[2]; };
union FragC  { v8f   v; float f[8]; };

__device__ __forceinline__ v8f splat8(float x) {
    v8f v = {x, x, x, x, x, x, x, x};
    return v;
}

__device__ __forceinline__ float fast_tanh(float x) {
#if __has_builtin(__builtin_amdgcn_tanhf)
    return __builtin_amdgcn_tanhf(x);     // V_TANH_F32 (TRANS, co-exec w/ WMMA)
#else
    return tanhf(x);
#endif
}

__device__ __forceinline__ v8f wmma_bf16(v16bf a, v16bf b, v8f c) {
    // (neg_a, A, neg_b, B, c_mod, C, reuse_a, reuse_b)
    return __builtin_amdgcn_wmma_f32_16x16x32_bf16(false, a, false, b,
                                                   (short)0, c, false, false);
}

// Load one 16x32 bf16 fragment (A or B operand) from row-major storage
// (LDS or global). Lane L: row resolved by caller,
// K chunk = {kk..kk+7} U {kk+16..kk+23}, kk = ks*32 + (lane>=16)*8.
__device__ __forceinline__ v16bf load_frag(const __bf16* base, int row,
                                           int rowStride, int kk) {
    FragAB f;
    const __bf16* p = base + row * rowStride + kk;
    f.q[0] = *(const uint4*)(p);        // 8 bf16: K = kk..kk+7
    f.q[1] = *(const uint4*)(p + 16);   // 8 bf16: K = kk+16..kk+23
    return f.v;
}

// fp32 -> bf16 weight conversion (one-time, tiny)
__global__ void cvt_f32_bf16(const float* __restrict__ src,
                             __bf16* __restrict__ dst, int n) {
    int i = blockIdx.x * blockDim.x + threadIdx.x;
    if (i < n) dst[i] = (__bf16)src[i];
}

__global__ void __launch_bounds__(NTHREADS)
fused_rnn_ssm(const float* __restrict__ pre_x, const float* __restrict__ pre_y,
              const float* __restrict__ fx,
              const float* __restrict__ A_b, const float* __restrict__ B_b,
              const float* __restrict__ C_b,
              const float* __restrict__ b_ih, const float* __restrict__ b_hh,
              const __bf16* __restrict__ Wih_bf, const __bf16* __restrict__ Whh_bf,
              const __bf16* __restrict__ Aw_bf,  const __bf16* __restrict__ Bw_bf,
              const __bf16* __restrict__ Cw_bf,
              float* __restrict__ out) {
    __shared__ __bf16 sXY[BM * 128];        // staged input tile (bf16)
    __shared__ __bf16 sH[2 * BM * HSTR];    // h double buffer (bf16)

    const int tid  = threadIdx.x;
    const int wave = tid >> 5;
    const int lane = tid & 31;
    const int half = lane >> 4;        // K-half selector for A/B fragments
    const int mr   = lane & 15;        // row within M-tile / col within N-tile
    const int bb   = blockIdx.x * BM;  // batch base for this block

    // This wave's state N-tile (16 output columns of the recurrence)
    const int nc = wave * 16 + mr;

    // Folded biases (accumulator init values)
    const float bias_e = b_ih[nc] + b_hh[nc];
    const float bias_s = A_b[nc] + B_b[nc];
    const int   ycol   = (wave & 3) * 16 + mr;            // waves 0..3 only
    const float ybias  = (wave < 4) ? C_b[ycol] : 0.0f;

    // ---------------- Phase 1: encoder RNN ----------------
    // Preload this wave's weight fragments into registers (loop-invariant).
    v16bf wih[4], whh[8];
    #pragma unroll
    for (int ks = 0; ks < 4; ++ks)
        wih[ks] = load_frag(Wih_bf, nc, 128, ks * 32 + half * 8);
    #pragma unroll
    for (int ks = 0; ks < 8; ++ks)
        whh[ks] = load_frag(Whh_bf, nc, 256, ks * 32 + half * 8);

    for (int i = tid; i < 2 * BM * HSTR; i += NTHREADS) sH[i] = (__bf16)0.0f;
    __syncthreads();

    // Per-thread input-tile loader coords (16 x 128, 4 elems/thread)
    const int er = (tid * 4) >> 7;
    const int ec = (tid * 4) & 127;

    #pragma unroll 1
    for (int t = 0; t < T_PRE; ++t) {
        // stage xy_t tile -> LDS (bf16)
        {
            const size_t rowb = (size_t)t * BATCH + bb + er;
            const float* s = (ec < K_IN) ? (pre_x + rowb * K_IN + ec)
                                         : (pre_y + rowb * K_OUT + (ec - K_IN));
            float4 f0 = *(const float4*)(s);
            union { __bf16 h[4]; uint2 q; } u;
            u.h[0] = (__bf16)f0.x; u.h[1] = (__bf16)f0.y;
            u.h[2] = (__bf16)f0.z; u.h[3] = (__bf16)f0.w;
            *(uint2*)(sXY + er * 128 + ec) = u.q;
            if (t + 1 < T_PRE) {
                const size_t nb = (size_t)(t + 1) * BATCH + bb + er;
                const float* pf = (ec < K_IN) ? (pre_x + nb * K_IN + ec)
                                              : (pre_y + nb * K_OUT + (ec - K_IN));
                __builtin_prefetch(pf, 0, 0);
            }
        }
        __syncthreads();

        const int cur = t & 1, nxt = cur ^ 1;
        const __bf16* hc = sH + cur * (BM * HSTR);

        v8f acc = splat8(bias_e);
        #pragma unroll
        for (int ks = 0; ks < 4; ++ks) {            // input proj, K = 128
            v16bf a = load_frag(sXY, mr, 128, ks * 32 + half * 8);
            acc = wmma_bf16(a, wih[ks], acc);
        }
        #pragma unroll
        for (int ks = 0; ks < 8; ++ks) {            // recurrence, K = 256
            v16bf a = load_frag(hc, mr, HSTR, ks * 32 + half * 8);
            acc = wmma_bf16(a, whh[ks], acc);
        }

        // h_new = tanh(acc) -> LDS (bf16), C/D layout: VGPR r -> M = half*8+r
        {
            FragC c0; c0.v = acc;
            __bf16* hn = sH + nxt * (BM * HSTR);
            #pragma unroll
            for (int r = 0; r < 8; ++r)
                hn[(half * 8 + r) * HSTR + nc] = (__bf16)fast_tanh(c0.f[r]);
        }
        __syncthreads();
    }
    // final h is in buffer 0 (T_PRE even)

    // ---------------- Phase 2: state-space rollout ----------------
    v16bf aw[8], bw[2], cw[8];
    #pragma unroll
    for (int ks = 0; ks < 8; ++ks)
        aw[ks] = load_frag(Aw_bf, nc, 256, ks * 32 + half * 8);
    #pragma unroll
    for (int ks = 0; ks < 2; ++ks)
        bw[ks] = load_frag(Bw_bf, nc, 64, ks * 32 + half * 8);
    if (wave < 4) {
        #pragma unroll
        for (int ks = 0; ks < 8; ++ks)
            cw[ks] = load_frag(Cw_bf, ycol, 256, ks * 32 + half * 8);
    }

    const int fr = (tid * 2) >> 6;   // 16 x 64 tile, 2 elems/thread
    const int fc = (tid * 2) & 63;

    #pragma unroll 1
    for (int t = 0; t < T_FWD; ++t) {
        // stage x_t tile -> LDS (bf16)
        {
            const float* s = fx + ((size_t)t * BATCH + bb + fr) * K_IN + fc;
            float2 f0 = *(const float2*)(s);
            union { __bf16 h[2]; unsigned int q; } u;
            u.h[0] = (__bf16)f0.x; u.h[1] = (__bf16)f0.y;
            *(unsigned int*)(sXY + fr * 64 + fc) = u.q;
            if (t + 1 < T_FWD)
                __builtin_prefetch(fx + ((size_t)(t + 1) * BATCH + bb + fr) * K_IN + fc, 0, 0);
        }
        __syncthreads();

        const int cur = t & 1, nxt = cur ^ 1;
        const __bf16* hc = sH + cur * (BM * HSTR);
        __bf16* hn = sH + nxt * (BM * HSTR);

        v8f acc = splat8(bias_s);
        #pragma unroll
        for (int ks = 0; ks < 2; ++ks) {            // x @ B_w.T, K = 64
            v16bf a = load_frag(sXY, mr, 64, ks * 32 + half * 8);
            acc = wmma_bf16(a, bw[ks], acc);
        }
        #pragma unroll
        for (int ks = 0; ks < 8; ++ks) {            // h @ A_w.T, K = 256
            v16bf a = load_frag(hc, mr, HSTR, ks * 32 + half * 8);
            acc = wmma_bf16(a, aw[ks], acc);
        }

        // h_new -> LDS (bf16)
        {
            FragC c0; c0.v = acc;
            #pragma unroll
            for (int r = 0; r < 8; ++r)
                hn[(half * 8 + r) * HSTR + nc] = (__bf16)c0.f[r];
        }
        __syncthreads();

        // y_t = h_new @ C_w.T + C_b : 4 output N-tiles, waves 0..3
        if (wave < 4) {
            v8f yacc = splat8(ybias);
            #pragma unroll
            for (int ks = 0; ks < 8; ++ks) {
                v16bf a = load_frag(hn, mr, HSTR, ks * 32 + half * 8);
                yacc = wmma_bf16(a, cw[ks], yacc);
            }
            FragC cy; cy.v = yacc;
            #pragma unroll
            for (int r = 0; r < 8; ++r)
                out[((size_t)t * BATCH + bb + half * 8 + r) * K_OUT + ycol] = cy.f[r];
        }
    }
}

extern "C" void kernel_launch(void* const* d_in, const int* in_sizes, int n_in,
                              void* d_out, int out_size, void* d_ws, size_t ws_size,
                              hipStream_t stream) {
    (void)in_sizes; (void)n_in; (void)out_size; (void)ws_size;
    const float* pre_x = (const float*)d_in[0];
    const float* pre_y = (const float*)d_in[1];
    const float* fx    = (const float*)d_in[2];
    const float* A_w   = (const float*)d_in[3];
    const float* A_b   = (const float*)d_in[4];
    const float* B_w   = (const float*)d_in[5];
    const float* B_b   = (const float*)d_in[6];
    const float* C_w   = (const float*)d_in[7];
    const float* C_b   = (const float*)d_in[8];
    const float* W_ih  = (const float*)d_in[9];
    const float* b_ih  = (const float*)d_in[10];
    const float* W_hh  = (const float*)d_in[11];
    const float* b_hh  = (const float*)d_in[12];

    // workspace: bf16 copies of the 5 weight matrices (L2-resident, 384KB)
    char* ws = (char*)d_ws;
    __bf16* Wih_bf = (__bf16*)(ws + 0);        // 256x128 -> 64KB
    __bf16* Whh_bf = (__bf16*)(ws + 65536);    // 256x256 -> 128KB
    __bf16* Aw_bf  = (__bf16*)(ws + 196608);   // 256x256 -> 128KB
    __bf16* Bw_bf  = (__bf16*)(ws + 327680);   // 256x64  -> 32KB
    __bf16* Cw_bf  = (__bf16*)(ws + 360448);   // 64x256  -> 32KB

    const int cvtB = 256;
    cvt_f32_bf16<<<(K_STATE * (K_IN + K_OUT) + cvtB - 1) / cvtB, cvtB, 0, stream>>>(W_ih, Wih_bf, K_STATE * (K_IN + K_OUT));
    cvt_f32_bf16<<<(K_STATE * K_STATE + cvtB - 1) / cvtB, cvtB, 0, stream>>>(W_hh, Whh_bf, K_STATE * K_STATE);
    cvt_f32_bf16<<<(K_STATE * K_STATE + cvtB - 1) / cvtB, cvtB, 0, stream>>>(A_w, Aw_bf, K_STATE * K_STATE);
    cvt_f32_bf16<<<(K_STATE * K_IN + cvtB - 1) / cvtB, cvtB, 0, stream>>>(B_w, Bw_bf, K_STATE * K_IN);
    cvt_f32_bf16<<<(K_OUT * K_STATE + cvtB - 1) / cvtB, cvtB, 0, stream>>>(C_w, Cw_bf, K_OUT * K_STATE);

    dim3 grid(BATCH / BM);
    fused_rnn_ssm<<<grid, NTHREADS, 0, stream>>>(
        pre_x, pre_y, fx, A_b, B_b, C_b, b_ih, b_hh,
        Wih_bf, Whh_bf, Aw_bf, Bw_bf, Cw_bf, (float*)d_out);
}
